// RBFN_CNF_30915174597279
// MI455X (gfx1250) — compile-verified, matrix operand order checked by reading, not run on hardware
//
#include <hip/hip_runtime.h>
#include <hip/hip_bf16.h>

typedef __attribute__((ext_vector_type(2)))  float  v2f;
typedef __attribute__((ext_vector_type(4)))  float  v4f;
typedef __attribute__((ext_vector_type(8)))  float  v8f;
typedef __attribute__((ext_vector_type(16))) __bf16 v16bf;

#define B_   16384
#define D_   64
#define NT_  128
#define NZ_  2048
#define WSTR (NZ_ + 1)   // 2049

// workspace layout (float offsets)
#define OFF_WT    0
#define OFF_CA    (OFF_WT  + D_*WSTR)        // centres, f32 A-operand swizzle
#define OFF_WTA   (OFF_CA  + NZ_*D_)         // Wt,      f32 A-operand swizzle
#define OFF_WTB   (OFF_WTA + NZ_*D_)         // Wt^T,    bf16 B-operand swizzle (NZ_*D_ halves)
#define OFF_CN    (OFF_WTB + NZ_*D_/2)
#define OFF_IS2   (OFF_CN  + NZ_)
#define OFF_Q     (OFF_IS2 + NZ_)

// ---------------------------------------------------------------------------
// Kernel 1: phi_t + W_t = einsum('dij,i->dj', W, phi_t)   (streams the 67MB W)
// ---------------------------------------------------------------------------
__global__ __launch_bounds__(256) void rbfn_prep_wt(
    const float* __restrict__ t, const float* __restrict__ ct,
    const float* __restrict__ lst, const float* __restrict__ W,
    float* __restrict__ Wt)
{
    __shared__ float ph[NT_];
    const int tid = threadIdx.x;
    if (tid < NT_) {
        float r = fabsf(t[0] - ct[tid]) / __expf(lst[tid]);
        ph[tid] = __expf(-r * r);
    }
    __syncthreads();
    const int d = blockIdx.x / 9;
    const int j = (blockIdx.x % 9) * 256 + tid;
    if (j < WSTR) {
        float acc = 0.f;
        const float* wp = W + (size_t)d * NT_ * WSTR + j;
        #pragma unroll 4
        for (int i = 0; i < NT_; ++i) acc += wp[(size_t)i * WSTR] * ph[i];
        Wt[d * WSTR + j] = acc;
    }
}

// ---------------------------------------------------------------------------
// Kernel 2: pre-swizzle operands into exact WMMA register layouts + meta
//   f32 A 16x4 layout: lane = M(0..15), half selects K{0,1}/{2,3}, vgpr = K lsb
//   bf16 B 32x16 layout: lane = N, vgpr v pairs: K = base(8*half) + {2v..} / 16+..
// ---------------------------------------------------------------------------
__global__ __launch_bounds__(256) void rbfn_prep_swz(
    const float* __restrict__ cz, const float* __restrict__ lsz,
    const float* __restrict__ Wt,
    float* __restrict__ cA, float* __restrict__ WtA, __bf16* __restrict__ WtB,
    float* __restrict__ cn, float* __restrict__ is2v, float* __restrict__ qv)
{
    const int tid = blockIdx.x * 256 + threadIdx.x;
    const int N0 = NZ_ * 64;        // cA elements
    const int N1 = N0 + NZ_ * 64;   // WtA elements
    const int N2 = N1 + NZ_ * 64;   // WtB halves
    const int N3 = N2 + NZ_;        // meta
    if (tid < N0) {
        int o = tid, c16 = o >> 10, rem = o & 1023, lane = rem >> 5, idx = rem & 31;
        int j   = c16 * 16 + (lane & 15);
        int col = 4 * (idx >> 1) + 2 * (lane >> 4) + (idx & 1);
        cA[o] = cz[j * 64 + col];
    } else if (tid < N1) {
        int o = tid - N0, c16 = o >> 10, rem = o & 1023, lane = rem >> 5, idx = rem & 31;
        int j   = c16 * 16 + (lane & 15);
        int col = 4 * (idx >> 1) + 2 * (lane >> 4) + (idx & 1);
        WtA[o] = Wt[col * WSTR + j];
    } else if (tid < N2) {
        int o = tid - N1;
        int cc = o >> 11, rem = o & 2047, dt = rem >> 9, r2 = rem & 511;
        int lane = r2 >> 4, hh = r2 & 15;
        int base = (lane >= 16) ? 8 : 0;
        int v = hh >> 1, e = hh & 1;
        int K = (v < 4) ? (base + 2 * v + e) : (16 + base + 2 * (v - 4) + e);
        int d = dt * 16 + (lane & 15);
        int j = cc * 32 + K;
        WtB[o] = (__bf16)Wt[d * WSTR + j];
    } else if (tid < N3) {
        int j = tid - N2;
        float c2 = 0.f, qq = 0.f;
        for (int d = 0; d < 64; ++d) {
            float c = cz[j * 64 + d];
            c2 += c * c;
            qq += Wt[d * WSTR + j] * c;
        }
        cn[j]   = c2;
        is2v[j] = __expf(-2.f * lsz[j]);
        qv[j]   = qq;
    }
}

// ---------------------------------------------------------------------------
// Kernel 3: fused main. One wave = one 16-row z tile, held in registers as the
// shared f32 B operand. Transposed fp32 WMMA tiles put phi directly into the
// bf16 A-operand lane layout — no LDS transpose.
// ---------------------------------------------------------------------------
__global__ __launch_bounds__(256) void rbfn_main(
    const float* __restrict__ z,
    const float* __restrict__ cA, const float* __restrict__ WtA,
    const __bf16* __restrict__ WtB, const float* __restrict__ Wt,
    const float* __restrict__ cn, const float* __restrict__ is2,
    const float* __restrict__ q,
    float* __restrict__ dz_out, float* __restrict__ dlogp_out)
{
    const int lane = threadIdx.x & 31;
    const int wave = threadIdx.x >> 5;
    const int row0 = (blockIdx.x * 8 + wave) * 16;
    const int m    = lane & 15;
    const int h    = lane >> 4;

    // z tile as f32 WMMA B operand: step s holds K = {4s+2h, 4s+2h+1}, N = m
    v2f zb[16];
    const float* zrow = z + (size_t)(row0 + m) * D_;
    #pragma unroll
    for (int s = 0; s < 16; ++s)
        zb[s] = *(const v2f*)(zrow + 4 * s + 2 * h);

    float zn_part = 0.f;
    #pragma unroll
    for (int s = 0; s < 16; ++s) zn_part += zb[s].x * zb[s].x + zb[s].y * zb[s].y;
    const float znorm = zn_part + __shfl_xor(zn_part, 16, 32);

    v8f dz[4] = {v8f{}, v8f{}, v8f{}, v8f{}};
    float tr = 0.f;

    for (int cc = 0; cc < NZ_ / 32; ++cc) {
        float phiA[8], phiB[8];
        #pragma unroll
        for (int sub = 0; sub < 2; ++sub) {
            const int c16 = cc * 2 + sub;
            const v2f* pA = (const v2f*)(cA  + (size_t)(c16 * 32 + lane) * 32);
            const v2f* pW = (const v2f*)(WtA + (size_t)(c16 * 32 + lane) * 32);
            v8f dot = {};   // (C z^T)[j, m]
            v8f zw  = {};   // (Wz z^T)[j, m]
            #pragma unroll
            for (int s = 0; s < 16; ++s) {
                dot = __builtin_amdgcn_wmma_f32_16x16x4_f32(false, pA[s], false, zb[s],
                                                            (short)0, dot, false, false);
                zw  = __builtin_amdgcn_wmma_f32_16x16x4_f32(false, pW[s], false, zb[s],
                                                            (short)0, zw,  false, false);
            }
            const int jb = c16 * 16 + 8 * h;     // vgpr r -> centre j = jb + r
            v4f cn0 = *(const v4f*)(cn  + jb), cn1 = *(const v4f*)(cn  + jb + 4);
            v4f s20 = *(const v4f*)(is2 + jb), s21 = *(const v4f*)(is2 + jb + 4);
            v4f q0  = *(const v4f*)(q   + jb), q1  = *(const v4f*)(q   + jb + 4);
            float* phis = sub ? phiB : phiA;
            #pragma unroll
            for (int r = 0; r < 8; ++r) {
                float cnr = (r < 4) ? cn0[r] : cn1[r - 4];
                float s2r = (r < 4) ? s20[r] : s21[r - 4];
                float qr  = (r < 4) ? q0[r]  : q1[r - 4];
                float sq  = fmaxf(znorm + cnr - 2.f * dot[r], 0.f);
                float ph  = __expf(-sq * s2r);
                phis[r] = ph;
                tr += ph * s2r * (zw[r] - qr);   // dlogp = 2 * sum
            }
        }
        // phi fragments are already in bf16 A(16x32) lane order: just pack.
        v16bf a;
        #pragma unroll
        for (int i = 0; i < 8; ++i) { a[i] = (__bf16)phiA[i]; a[8 + i] = (__bf16)phiB[i]; }

        const __bf16* pB = WtB + (size_t)(cc * 4 * 32 + lane) * 16;
        #pragma unroll
        for (int dt = 0; dt < 4; ++dt) {
            v16bf b = *(const v16bf*)(pB + (size_t)dt * 32 * 16);
            dz[dt] = __builtin_amdgcn_wmma_f32_16x16x32_bf16(false, a, false, b,
                                                             (short)0, dz[dt], false, false);
        }
    }

    // epilogue: bias add + store dz_dt (row = r + 8h, col = dt*16 + m)
    #pragma unroll
    for (int dt = 0; dt < 4; ++dt) {
        const float bias = Wt[(dt * 16 + m) * WSTR + NZ_];
        #pragma unroll
        for (int r = 0; r < 8; ++r) {
            const int row = row0 + r + 8 * h;
            dz_out[(size_t)row * D_ + dt * 16 + m] = dz[dt][r] + bias;
        }
    }
    const float trt = tr + __shfl_xor(tr, 16, 32);
    if (h == 0) dlogp_out[row0 + m] = 2.0f * trt;
}

// ---------------------------------------------------------------------------
extern "C" void kernel_launch(void* const* d_in, const int* in_sizes, int n_in,
                              void* d_out, int out_size, void* d_ws, size_t ws_size,
                              hipStream_t stream) {
    const float* t   = (const float*)d_in[0];
    const float* z   = (const float*)d_in[1];
    const float* cz  = (const float*)d_in[3];
    const float* lsz = (const float*)d_in[4];
    const float* ct  = (const float*)d_in[5];
    const float* lst = (const float*)d_in[6];
    const float* W   = (const float*)d_in[7];

    float*  ws  = (float*)d_ws;
    float*  Wt  = ws + OFF_WT;
    float*  cA  = ws + OFF_CA;
    float*  WtA = ws + OFF_WTA;
    __bf16* WtB = (__bf16*)(ws + OFF_WTB);
    float*  cn  = ws + OFF_CN;
    float*  is2 = ws + OFF_IS2;
    float*  q   = ws + OFF_Q;

    float* dz_out = (float*)d_out;
    float* lp_out = dz_out + (size_t)B_ * D_;

    rbfn_prep_wt <<<dim3(64 * 9), 256, 0, stream>>>(t, ct, lst, W, Wt);
    rbfn_prep_swz<<<dim3(1544),   256, 0, stream>>>(cz, lsz, Wt, cA, WtA, WtB, cn, is2, q);
    rbfn_main    <<<dim3(B_ / 128), 256, 0, stream>>>(z, cA, WtA, WtB, Wt, cn, is2, q,
                                                      dz_out, lp_out);
}
